// FFTLanguageModel_86904368267650
// MI455X (gfx1250) — matrix operand order repaired and checked by reading, not compiled, not run on hardware
//
#include <hip/hip_runtime.h>
#include <hip/hip_bf16.h>

// Problem dims (FFT language model forward)
#define VV   8192
#define SS   128
#define HH   256
#define FF   65
#define HF   (HH * FF)      // 16640 elements per token
#define HF4  (HF / 4)       // 4160 float4 chunks
#define BT   (16 * SS)      // 2048 tokens total
#define LN_EPS 1e-5f

typedef __attribute__((ext_vector_type(16))) _Float16 v16h;
typedef __attribute__((ext_vector_type(8)))  _Float16 v8h;
typedef __attribute__((ext_vector_type(8)))  float    v8f;

// ---------------------------------------------------------------------------
// Kernel 1: per-token gather + |z| + LayerNorm + Re-projection -> features f16
// One workgroup (256 threads = 8 wave32) per token. mag & cos(phase) staged in
// 133 KB of dynamic LDS (CDNA5 WGP: 320 KB) so HBM is touched exactly once.
// Pass 1 is fully float4-vectorized: row base = token*66560 B (16B aligned),
// chunk index * 16 B -> global_load_b128 / ds_store_b128 on all hot streams.
// ---------------------------------------------------------------------------
__global__ __launch_bounds__(256) void fftlm_features_kernel(
    const int*   __restrict__ x,
    const float* __restrict__ tok_re,
    const float* __restrict__ tok_im,
    const float* __restrict__ pos0_re,
    const float* __restrict__ pos0_im,
    const float* __restrict__ gamma,
    const float* __restrict__ beta,
    _Float16*    __restrict__ featH)   // [BT][HH]
{
    extern __shared__ float smem[];
    float* sMag = smem;        // HF floats
    float* sCos = smem + HF;   // HF floats
    __shared__ float sRed[2][8];
    __shared__ float sStats[2];

    const int  t    = blockIdx.x;
    const int  tid  = threadIdx.x;
    const long base = (long)x[t] * (long)HF;

    const float4* re4 = (const float4*)(tok_re + base);
    const float4* im4 = (const float4*)(tok_im + base);
    const float4* pr4 = (const float4*)pos0_re;
    const float4* pi4 = (const float4*)pos0_im;

    // Pass 1: stream 2*16640 f32 from HBM as b128 loads, build mag/cos in LDS,
    // accumulate sum and sum-of-squares for the joint LayerNorm.
    float sum = 0.f, sq = 0.f;
    for (int v = tid; v < HF4; v += 256) {
        float4 re = re4[v], im = im4[v];
        float4 pr = pr4[v], pi = pi4[v];
        re.x += pr.x; re.y += pr.y; re.z += pr.z; re.w += pr.w;
        im.x += pi.x; im.y += pi.y; im.z += pi.z; im.w += pi.w;

        float4 mg, cs;
        {
            float m2;
            m2 = re.x*re.x + im.x*im.x; mg.x = sqrtf(m2); sq += m2;
            cs.x = (mg.x > 0.f) ? (re.x / mg.x) : 1.0f;   // cos(atan2)=re/|z|
            m2 = re.y*re.y + im.y*im.y; mg.y = sqrtf(m2); sq += m2;
            cs.y = (mg.y > 0.f) ? (re.y / mg.y) : 1.0f;
            m2 = re.z*re.z + im.z*im.z; mg.z = sqrtf(m2); sq += m2;
            cs.z = (mg.z > 0.f) ? (re.z / mg.z) : 1.0f;
            m2 = re.w*re.w + im.w*im.w; mg.w = sqrtf(m2); sq += m2;
            cs.w = (mg.w > 0.f) ? (re.w / mg.w) : 1.0f;
        }
        sum += mg.x + mg.y + mg.z + mg.w;
        *(float4*)(sMag + v * 4) = mg;   // ds_store_b128
        *(float4*)(sCos + v * 4) = cs;   // ds_store_b128
    }

    // wave32 tree reduce, then cross-wave via LDS
    #pragma unroll
    for (int o = 16; o > 0; o >>= 1) {
        sum += __shfl_down(sum, o, 32);
        sq  += __shfl_down(sq,  o, 32);
    }
    const int lane = tid & 31, wid = tid >> 5;
    if (lane == 0) { sRed[0][wid] = sum; sRed[1][wid] = sq; }
    __syncthreads();
    if (tid == 0) {
        float s = 0.f, q = 0.f;
        #pragma unroll
        for (int w = 0; w < 8; ++w) { s += sRed[0][w]; q += sRed[1][w]; }
        const float mean = s * (1.0f / (float)HF);
        const float var  = q * (1.0f / (float)HF) - mean * mean;
        sStats[0] = mean;
        sStats[1] = rsqrtf(var + LN_EPS);
    }
    __syncthreads();
    const float mean = sStats[0], rstd = sStats[1];

    // Pass 2 (from LDS): thread h reduces its 65 freqs.
    // Address h*65+f: stride 65 mod 64 banks == 1 -> conflict-free per wave.
    const int h   = tid;
    const int off = h * FF;
    float acc = 0.f;
    for (int f = 0; f < FF; ++f) {
        float nm = (sMag[off + f] - mean) * rstd * gamma[off + f] + beta[off + f];
        acc += nm * sCos[off + f];
    }
    featH[(long)t * HH + h] = (_Float16)acc;
}

// ---------------------------------------------------------------------------
// Kernel 2: W [V,H] f32 -> f16 (row-major preserved: K=h stays contiguous,
// which is exactly the B-fragment layout WMMA wants — no transpose needed).
// ---------------------------------------------------------------------------
__global__ __launch_bounds__(256) void fftlm_wconv_kernel(
    const float* __restrict__ W, _Float16* __restrict__ Wh)
{
    const int i = (blockIdx.x * 256 + threadIdx.x) * 4;  // VV*HH divisible by 1024
    float4 w = *(const float4*)(W + i);
    union { _Float16 h[4]; uint2 u; } o;
    o.h[0] = (_Float16)w.x; o.h[1] = (_Float16)w.y;
    o.h[2] = (_Float16)w.z; o.h[3] = (_Float16)w.w;
    *(uint2*)(Wh + i) = o.u;
}

// ---------------------------------------------------------------------------
// Kernel 3: logits[2048,8192] = features[2048,256] * W^T + b via WMMA f16.
// One wave -> one 16x16 C tile, K=256 -> 8x v_wmma_f32_16x16x32_f16.
// Fragment layouts per CDNA5 ISA 7.12.2 (wave32):
//   A (16x32 f16): lanes 0-15 hold M=lane, K {kb*32+0..7, +16..23};
//                  lanes 16-31 hold M=lane-16, K {+8..15, +24..31}.
//   B (32x16 f16): lane n = lane&15 holds K in the same order as A
//                  (lanes 0-15: K 0..7, 16..23; lanes 16-31: K 8..15, 24..31).
//   C/D (16x16 f32): lane -> N = lane&15; VGPR r -> M = r + (lane>=16 ? 8:0).
// Logits are streamed out with non-temporal stores (no reuse; keep L2 for the
// embedding gather + W tiles).
// ---------------------------------------------------------------------------
__global__ __launch_bounds__(256) void fftlm_gemm_kernel(
    const _Float16* __restrict__ A,    // featH [BT][HH]
    const _Float16* __restrict__ Bm,   // Wh    [VV][HH]
    const float*    __restrict__ bias, // [VV]
    float*          __restrict__ out)  // [BT][VV]
{
    const int lane = threadIdx.x & 31;
    const int wave = threadIdx.x >> 5;
    const int mt   = blockIdx.x >> 6;            // 0..127  (2048 rows / 16)
    const int ntg  = blockIdx.x & 63;            // 0..63
    const int nt   = ntg * 8 + wave;             // 0..511  (8192 cols / 16)

    const int hi     = lane >> 4;                // 0 or 1 (half-wave)
    const int mrow   = mt * 16 + (lane & 15);
    const int nrowv  = nt * 16 + (lane & 15);
    const int khalf  = hi ? 8  : 0;              // A interleave
    const int kbase  = hi ? 16 : 0;              // unused (B uses khalf)

    const _Float16* aptr = A  + (long)mrow  * HH;
    const _Float16* bptr = Bm + (long)nrowv * HH;

    v8f acc = {};
    #pragma unroll
    for (int kb = 0; kb < 8; ++kb) {
        const int k0 = kb * 32;
        if (kb < 7) __builtin_prefetch(bptr + k0 + 32, 0, 1);  // global_prefetch_b8

        union { v16h v; v8h h[2]; } a;
        a.h[0] = *(const v8h*)(aptr + k0 + khalf);         // K k0+khalf .. +7
        a.h[1] = *(const v8h*)(aptr + k0 + 16 + khalf);    // K k0+16+khalf .. +7

        union { v16h v; v8h h[2]; } bf;
        bf.h[0] = *(const v8h*)(bptr + k0 + khalf);        // K k0+khalf .. +7
        bf.h[1] = *(const v8h*)(bptr + k0 + 16 + khalf);   // K k0+16+khalf .. +7

        acc = __builtin_amdgcn_wmma_f32_16x16x32_f16(
            /*neg_a=*/false, a.v, /*neg_b=*/false, bf.v,
            /*c_mod=*/(short)0, acc, /*reuse_a=*/false, /*reuse_b=*/false);
    }

    const int   n   = nt * 16 + (lane & 15);
    const float bv  = bias[n];
    float* orow = out + (long)(mt * 16 + (hi ? 8 : 0)) * VV + n;
    #pragma unroll
    for (int r = 0; r < 8; ++r)
        __builtin_nontemporal_store(acc[r] + bv, orow + (long)r * VV);
}

// ---------------------------------------------------------------------------
extern "C" void kernel_launch(void* const* d_in, const int* in_sizes, int n_in,
                              void* d_out, int out_size, void* d_ws, size_t ws_size,
                              hipStream_t stream)
{
    const int*   x       = (const int*)  d_in[0];
    const float* tok_re  = (const float*)d_in[1];
    const float* tok_im  = (const float*)d_in[2];
    const float* pos0_re = (const float*)d_in[3];
    const float* pos0_im = (const float*)d_in[4];
    const float* gamma   = (const float*)d_in[5];
    const float* beta    = (const float*)d_in[6];
    const float* W       = (const float*)d_in[7];
    const float* bias    = (const float*)d_in[8];
    float* out = (float*)d_out;

    // Workspace: featH (1 MB) then Wh (4 MB)
    _Float16* featH = (_Float16*)d_ws;
    _Float16* Wh    = (_Float16*)((char*)d_ws + (size_t)BT * HH * sizeof(_Float16));

    const size_t lds = (size_t)2 * HF * sizeof(float);  // 133,120 B < 320 KB WGP LDS
    fftlm_features_kernel<<<BT, 256, lds, stream>>>(
        x, tok_re, tok_im, pos0_re, pos0_im, gamma, beta, featH);

    fftlm_wconv_kernel<<<(VV * HH / 4) / 256, 256, 0, stream>>>(W, Wh);

    fftlm_gemm_kernel<<<128 * 64, 256, 0, stream>>>(featH, Wh, bias, out);
}
